// CharCNN_86346022519013
// MI455X (gfx1250) — matrix operand order repaired and checked by reading, not compiled, not run on hardware
//
#include <hip/hip_runtime.h>
#include <hip/hip_bf16.h>

// ---------------------------------------------------------------------------
// CharCNN for MI455X (gfx1250, wave32, WMMA)
//   Phase 0: convert lin_w f32 -> f16 (row-major [768, 4000] kept as-is)
//   Phase 1: embedding gather + grouped conv1d + maxpool + ReLU -> h[4096,4000] f16
//   Phase 2: out = h @ lin_w^T + b via v_wmma_f32_16x16x32_f16
// ---------------------------------------------------------------------------

typedef __attribute__((ext_vector_type(16))) _Float16 v16h;
typedef __attribute__((ext_vector_type(8)))  _Float16 v8h;
typedef __attribute__((ext_vector_type(8)))  float    v8f;

#define EMB_DIM   50
#define W_CHARS   20
#define HIDDEN    4000
#define OUT_DIM   768
#define NSAMP     4096          // B*S = 16*256
#define FEATS_PER_CFG 800       // EMB_DIM * 16

// ---------------------------------------------------------------------------
// Kernel 0: f32 -> f16 convert (grid-stride)
// ---------------------------------------------------------------------------
__global__ void __launch_bounds__(256)
cvt_f32_to_f16(const float* __restrict__ src, _Float16* __restrict__ dst, int n) {
    int i = blockIdx.x * blockDim.x + threadIdx.x;
    int stride = gridDim.x * blockDim.x;
    for (; i < n; i += stride) dst[i] = (_Float16)src[i];
}

// ---------------------------------------------------------------------------
// Kernel 1: char CNN features. One block per (b,s) sample.
//   xs[e][t] staged in LDS (50*20 floats = 4KB). Each thread computes a
//   strided subset of the 4000 features; conv loops fully unrolled per K.
// ---------------------------------------------------------------------------
template<int K>
__device__ __forceinline__ float conv_maxpool(const float* __restrict__ xe,
                                              const float* __restrict__ wk) {
    float m = -3.0e38f;
    #pragma unroll
    for (int t = 0; t <= W_CHARS - K; ++t) {
        float acc = 0.0f;
        #pragma unroll
        for (int i = 0; i < K; ++i) acc = fmaf(xe[t + i], wk[i], acc);
        m = fmaxf(m, acc);
    }
    return m;
}

__global__ void __launch_bounds__(256)
charcnn_feats(const int*   __restrict__ cid,   // [NSAMP, 20]
              const float* __restrict__ emb,   // [101, 50]
              const float* __restrict__ w1, const float* __restrict__ w2,
              const float* __restrict__ w3, const float* __restrict__ w4,
              const float* __restrict__ w5,
              const float* __restrict__ b1, const float* __restrict__ b2,
              const float* __restrict__ b3, const float* __restrict__ b4,
              const float* __restrict__ b5,
              _Float16*    __restrict__ h)     // [NSAMP, HIDDEN] f16
{
    __shared__ int   ids[W_CHARS];
    __shared__ float xs[EMB_DIM * W_CHARS];    // xs[e*20 + t]

    const int s   = blockIdx.x;
    const int tid = threadIdx.x;

    if (tid < W_CHARS) ids[tid] = cid[s * W_CHARS + tid];
    __syncthreads();

    for (int idx = tid; idx < EMB_DIM * W_CHARS; idx += blockDim.x) {
        const int e = idx / W_CHARS;
        const int t = idx - e * W_CHARS;
        xs[idx] = emb[ids[t] * EMB_DIM + e];
    }
    __syncthreads();

    const float* ws[5] = { w1, w2, w3, w4, w5 };
    const float* bs[5] = { b1, b2, b3, b4, b5 };

    for (int fi = tid; fi < HIDDEN; fi += blockDim.x) {
        const int c = fi / FEATS_PER_CFG;          // config 0..4, kernel size c+1
        const int r = fi - c * FEATS_PER_CFG;      // e*16 + j
        const int k = c + 1;
        const float* wk = ws[c] + r * k;           // weight row [k]
        const float* xe = xs + (r >> 4) * W_CHARS; // e = r/16
        float m;
        switch (c) {
            case 0:  m = conv_maxpool<1>(xe, wk); break;
            case 1:  m = conv_maxpool<2>(xe, wk); break;
            case 2:  m = conv_maxpool<3>(xe, wk); break;
            case 3:  m = conv_maxpool<4>(xe, wk); break;
            default: m = conv_maxpool<5>(xe, wk); break;
        }
        float v = m + bs[c][r];                    // bias after maxpool (constant shift)
        v = v > 0.0f ? v : 0.0f;                   // ReLU
        h[(size_t)s * HIDDEN + fi] = (_Float16)v;
    }
}

// ---------------------------------------------------------------------------
// Kernel 2: WMMA GEMM  out[4096,768] = h[4096,4000] * wh[768,4000]^T + bias
//   Wave tile: 16(M) x 64(N), 4x v8f accumulators, K step 32 (125 iters).
//   A layout (16-bit A 16x32): lane<16 -> K 0..15 of row lane ; lane>=16 -> K 16..31
//   B layout (K=32 x N=16):    lane<16 -> K 0..15 of col lane ; lane>=16 -> K 16..31
//   Both map to two contiguous 16B global loads per lane.
// ---------------------------------------------------------------------------
__device__ __forceinline__ v16h load16h(const _Float16* __restrict__ p0,
                                        const _Float16* __restrict__ p1) {
    v8h lo = *(const v8h*)p0;
    v8h hi = *(const v8h*)p1;
    return __builtin_shufflevector(lo, hi, 0,1,2,3,4,5,6,7,8,9,10,11,12,13,14,15);
}

__global__ void __launch_bounds__(256)
wmma_gemm_bias(const _Float16* __restrict__ h,    // [NSAMP, HIDDEN]
               const _Float16* __restrict__ wh,   // [OUT_DIM, HIDDEN]
               const float*    __restrict__ bias, // [OUT_DIM]
               float*          __restrict__ out)  // [NSAMP, OUT_DIM]
{
    const int lane  = threadIdx.x & 31;
    const int wave  = threadIdx.x >> 5;
    const int tile  = blockIdx.x * 8 + wave;      // 3072 wave-tiles total
    const int tilesN = OUT_DIM / 64;              // 12
    const int tm = tile / tilesN;                 // 0..255
    const int tn = tile - tm * tilesN;            // 0..11
    const int m0 = tm * 16;
    const int n0 = tn * 64;

    const int l15 = lane & 15;
    const int hi  = lane >> 4;                    // 0 or 1

    const _Float16* arow = h  + (size_t)(m0 + l15) * HIDDEN + hi * 16;
    const _Float16* b0p  = wh + (size_t)(n0      + l15) * HIDDEN + hi * 16;
    const _Float16* b1p  = wh + (size_t)(n0 + 16 + l15) * HIDDEN + hi * 16;
    const _Float16* b2p  = wh + (size_t)(n0 + 32 + l15) * HIDDEN + hi * 16;
    const _Float16* b3p  = wh + (size_t)(n0 + 48 + l15) * HIDDEN + hi * 16;

    v8f c0 = {}, c1 = {}, c2 = {}, c3 = {};

    for (int k0 = 0; k0 < HIDDEN; k0 += 32) {
        const v16h a  = load16h(arow + k0, arow + k0 + 8);
        const v16h b0 = load16h(b0p + k0,  b0p + k0 + 8);
        const v16h b1 = load16h(b1p + k0,  b1p + k0 + 8);
        const v16h b2 = load16h(b2p + k0,  b2p + k0 + 8);
        const v16h b3 = load16h(b3p + k0,  b3p + k0 + 8);
        c0 = __builtin_amdgcn_wmma_f32_16x16x32_f16(false, a, false, b0, (short)0, c0, false, false);
        c1 = __builtin_amdgcn_wmma_f32_16x16x32_f16(false, a, false, b1, (short)0, c1, false, false);
        c2 = __builtin_amdgcn_wmma_f32_16x16x32_f16(false, a, false, b2, (short)0, c2, false, false);
        c3 = __builtin_amdgcn_wmma_f32_16x16x32_f16(false, a, false, b3, (short)0, c3, false, false);
    }

    // Epilogue: C/D layout -> VGPR r holds (M = r + 8*hi, N = l15); add bias.
    const int col = n0 + l15;
    const float bb0 = bias[col];
    const float bb1 = bias[col + 16];
    const float bb2 = bias[col + 32];
    const float bb3 = bias[col + 48];
    const int rowbase = m0 + hi * 8;
    #pragma unroll
    for (int r = 0; r < 8; ++r) {
        float* o = out + (size_t)(rowbase + r) * OUT_DIM + col;
        o[0]  = c0[r] + bb0;
        o[16] = c1[r] + bb1;
        o[32] = c2[r] + bb2;
        o[48] = c3[r] + bb3;
    }
}

// ---------------------------------------------------------------------------
// Launch
// ---------------------------------------------------------------------------
extern "C" void kernel_launch(void* const* d_in, const int* in_sizes, int n_in,
                              void* d_out, int out_size, void* d_ws, size_t ws_size,
                              hipStream_t stream) {
    const int*   cid = (const int*)d_in[0];
    const float* emb = (const float*)d_in[1];

    const float *w[5], *b[5], *lw, *lb;
    if (n_in >= 14) {
        // tuples flattened into individual entries
        for (int i = 0; i < 5; ++i) w[i] = (const float*)d_in[2 + i];
        for (int i = 0; i < 5; ++i) b[i] = (const float*)d_in[7 + i];
        lw = (const float*)d_in[12];
        lb = (const float*)d_in[13];
    } else {
        // tuples concatenated: d_in[2]=conv_ws flat, d_in[3]=conv_bs flat
        const float* wsflat = (const float*)d_in[2];
        const float* bsflat = (const float*)d_in[3];
        size_t woff = 0, boff = 0;
        for (int i = 0; i < 5; ++i) {
            w[i] = wsflat + woff;  woff += (size_t)FEATS_PER_CFG * (i + 1);
            b[i] = bsflat + boff;  boff += FEATS_PER_CFG;
        }
        lw = (const float*)d_in[4];
        lb = (const float*)d_in[5];
    }

    float* out = (float*)d_out;

    // workspace layout: wh f16 [768*4000], then h f16 [4096*4000]
    _Float16* wh = (_Float16*)d_ws;
    _Float16* hb = (_Float16*)((char*)d_ws + (size_t)OUT_DIM * HIDDEN * sizeof(_Float16));

    // Phase 0: convert lin_w to f16
    const int nW = OUT_DIM * HIDDEN;  // 3,072,000
    cvt_f32_to_f16<<<2048, 256, 0, stream>>>(lw, wh, nW);

    // Phase 1: features (one block per sample)
    charcnn_feats<<<NSAMP, 256, 0, stream>>>(cid, emb,
                                             w[0], w[1], w[2], w[3], w[4],
                                             b[0], b[1], b[2], b[3], b[4],
                                             hb);

    // Phase 2: WMMA GEMM + bias
    // wave tiles: (4096/16) * (768/64) = 3072 -> 384 blocks x 8 waves
    wmma_gemm_bias<<<384, 256, 0, stream>>>(hb, wh, lb, out);
}
